// TransformerModel_8684423872847
// MI455X (gfx1250) — matrix operand order repaired and checked by reading, not compiled, not run on hardware
//
#include <hip/hip_runtime.h>
#include <math.h>

// ---------------------------------------------------------------------------
// Transformer encoder forward for MI455X (gfx1250, wave32, WMMA).
// L=6, H=16, D=1024, F=4096, V=32000, B=8, S=512.  All big GEMMs run through
// v_wmma_f32_16x16x32_bf16 with double-buffered LDS fed by
// global_load_async_to_lds_b128 (ASYNCcnt); attention is single-pass flash
// with K/V^T staged in LDS.
// ---------------------------------------------------------------------------

#define DD   1024
#define FF   4096
#define LL   6
#define BB   8
#define SS   512
#define HH   16
#define DH   64
#define MM   (BB * SS)   // 4096 token rows

typedef __attribute__((ext_vector_type(16))) __bf16 v16bf;
typedef __attribute__((ext_vector_type(8)))  float  v8f;

union FragU { uint4 u[2]; v16bf v; };

__device__ __forceinline__ v8f wmma_bf16(v16bf a, v16bf b, v8f c) {
  // (neg_a, A, neg_b, B, c_mod, C, reuse_a, reuse_b)
  return __builtin_amdgcn_wmma_f32_16x16x32_bf16(false, a, false, b, (short)0, c, false, false);
}

// Async copy of 16 bytes/lane from global (saddr + per-lane 32-bit offset)
// directly into LDS at the per-lane 32-bit LDS byte address (ASYNCcnt-tracked).
__device__ __forceinline__ void async_ld_b128(uint32_t lds_addr, const void* sbase,
                                              uint32_t voff) {
  asm volatile("global_load_async_to_lds_b128 %0, %1, %2"
               :: "v"(lds_addr), "v"(voff), "s"(sbase)
               : "memory");
}
__device__ __forceinline__ void wait_async0() {
  asm volatile("s_wait_asynccnt 0x0" ::: "memory");
}
// Generic shared pointer -> LDS byte address (aperture: the low 32 address bits).
__device__ __forceinline__ uint32_t lds_addr_of(const void* p) {
  return (uint32_t)(uintptr_t)p;
}

// A-matrix 16x32 fragment (row-major source, row = base+lane%16):
//   lanes 0-15 : v0-7 = K 0..15 ; lanes 16-31 : v0-7 = K 16..31
__device__ __forceinline__ v16bf ld_afrag(const __bf16* row_ptr, int hf) {
  FragU f;
  f.u[0] = *(const uint4*)(row_ptr + hf * 16);
  f.u[1] = *(const uint4*)(row_ptr + hf * 16 + 8);
  return f.v;
}

// B-matrix 32x16 fragment from W^T[N,K] (lane = out column n = base+lane%16):
//   lanes 0-15 : v0-7 = K 0..15 ; lanes 16-31 : v0-7 = K 16..31
__device__ __forceinline__ v16bf ld_bfrag(const __bf16* row_ptr, int hf) {
  FragU f;
  f.u[0] = *(const uint4*)(row_ptr + hf * 16);
  f.u[1] = *(const uint4*)(row_ptr + hf * 16 + 8);
  return f.v;
}

// ---------------------------------------------------------------------------
// Weight transpose + f32->bf16 convert: src[K,N] f32 -> dst[N,K] bf16
// ---------------------------------------------------------------------------
__global__ __launch_bounds__(256) void transpose_cvt(const float* __restrict__ src,
                                                     __bf16* __restrict__ dst,
                                                     int K, int N) {
  __shared__ float tile[32][33];
  const int nb = blockIdx.x * 32, kb = blockIdx.y * 32;
  const int tx = threadIdx.x, ty = threadIdx.y;        // 32 x 8
#pragma unroll
  for (int i = 0; i < 4; ++i)
    tile[ty + 8 * i][tx] = src[(size_t)(kb + ty + 8 * i) * N + nb + tx];
  __syncthreads();
#pragma unroll
  for (int i = 0; i < 4; ++i)
    dst[(size_t)(nb + ty + 8 * i) * K + kb + tx] = (__bf16)tile[tx][ty + 8 * i];
}

// ---------------------------------------------------------------------------
// Block reduction helper (sum pair) -- 256 threads, 8 waves of 32
// ---------------------------------------------------------------------------
__device__ __forceinline__ void block_reduce2(float& a, float& b) {
#pragma unroll
  for (int m = 16; m; m >>= 1) {
    a += __shfl_xor(a, m, 32);
    b += __shfl_xor(b, m, 32);
  }
  __shared__ float sa[8], sb[8];
  const int wid = threadIdx.x >> 5, ln = threadIdx.x & 31;
  if (ln == 0) { sa[wid] = a; sb[wid] = b; }
  __syncthreads();
  a = 0.0f; b = 0.0f;
#pragma unroll
  for (int i = 0; i < 8; ++i) { a += sa[i]; b += sb[i]; }
  __syncthreads();
}

// ---------------------------------------------------------------------------
// Embedding + LayerNorm + length mask.  One block per token row.
// ---------------------------------------------------------------------------
__global__ __launch_bounds__(256) void embed_ln(const int* __restrict__ x,
                                                const int* __restrict__ lengths,
                                                const float* __restrict__ wemb,
                                                const float* __restrict__ pemb,
                                                const float* __restrict__ g,
                                                const float* __restrict__ bn,
                                                float* __restrict__ h,
                                                __bf16* __restrict__ hb) {
  const int row = blockIdx.x;
  const int b = row / SS, s = row % SS;
  const size_t tok = (size_t)x[row];
  const int valid = (s < lengths[b]) ? 1 : 0;
  float val[4], sum = 0.0f, sq = 0.0f;
#pragma unroll
  for (int j = 0; j < 4; ++j) {
    const int i = j * 256 + threadIdx.x;
    const float t = wemb[tok * DD + i] + pemb[(size_t)s * DD + i];
    val[j] = t; sum += t; sq += t * t;
  }
  block_reduce2(sum, sq);
  const float mean = sum * (1.0f / DD);
  const float var  = sq * (1.0f / DD) - mean * mean;
  const float rstd = rsqrtf(var + 1e-12f);
#pragma unroll
  for (int j = 0; j < 4; ++j) {
    const int i = j * 256 + threadIdx.x;
    float o = (val[j] - mean) * rstd * g[i] + bn[i];
    o = valid ? o : 0.0f;
    h[(size_t)row * DD + i]  = o;
    hb[(size_t)row * DD + i] = (__bf16)o;
  }
}

// ---------------------------------------------------------------------------
// residual + bias + LayerNorm (+ optional length mask). One block per row.
// ---------------------------------------------------------------------------
__global__ __launch_bounds__(256) void ln_residual(const float* __restrict__ hin,
                                                   const float* __restrict__ raw,
                                                   const float* __restrict__ bias,
                                                   const float* __restrict__ g,
                                                   const float* __restrict__ bn,
                                                   const int* __restrict__ lengths,
                                                   int maskflag,
                                                   float* __restrict__ hout,
                                                   __bf16* __restrict__ hb) {
  const int row = blockIdx.x;
  const int b = row / SS, s = row % SS;
  const int valid = (s < lengths[b]) ? 1 : 0;
  float val[4], sum = 0.0f, sq = 0.0f;
#pragma unroll
  for (int j = 0; j < 4; ++j) {
    const int i = j * 256 + threadIdx.x;
    const float t = hin[(size_t)row * DD + i] + raw[(size_t)row * DD + i] + bias[i];
    val[j] = t; sum += t; sq += t * t;
  }
  block_reduce2(sum, sq);
  const float mean = sum * (1.0f / DD);
  const float var  = sq * (1.0f / DD) - mean * mean;
  const float rstd = rsqrtf(var + 1e-12f);
#pragma unroll
  for (int j = 0; j < 4; ++j) {
    const int i = j * 256 + threadIdx.x;
    float o = (val[j] - mean) * rstd * g[i] + bn[i];
    if (maskflag && !valid) o = 0.0f;
    hout[(size_t)row * DD + i] = o;
    hb[(size_t)row * DD + i]   = (__bf16)o;
  }
}

// ---------------------------------------------------------------------------
// WMMA GEMM (NT):  C[M,N] = A[M,K](bf16) * Bt[N,K](bf16)^T
// Workgroup: 256 thr = 8 waves, 64x256 output tile; wave tile 32x64 (2x4 WMMA
// accumulators).  K staged in 32-deep double-buffered LDS tiles filled by
// global_load_async_to_lds_b128; one barrier per K stage.  The K loop is
// unrolled by 2 so the buffer parity is a compile-time constant (no indexed
// VGPR moves / cndmask in the issue path).
// mode 0: write raw f32            (bias unused)
// mode 1: (acc+bias)*scale -> bf16 (Q/K/V projections)
// mode 2: fast_gelu(acc+bias) -> bf16 (FFN up-proj)
// ---------------------------------------------------------------------------
#define LDSTR 40   // padded K stride (elements) against LDS bank conflicts

__device__ __forceinline__ void gemm_issue(uint32_t aL, const uint32_t* bL,
                                           const __bf16* A, const __bf16* Bt,
                                           uint32_t voffA, const uint32_t* voffB) {
  async_ld_b128(aL, (const void*)A, voffA);
#pragma unroll
  for (int j = 0; j < 4; ++j) async_ld_b128(bL[j], (const void*)Bt, voffB[j]);
}

__device__ __forceinline__ void gemm_stage(const __bf16 (*Asb)[LDSTR],
                                           const __bf16 (*Bsb)[LDSTR],
                                           int wm, int wn, int lm, int hf,
                                           v8f acc[2][4]) {
  v16bf af[2], bf[4];
#pragma unroll
  for (int i = 0; i < 2; ++i) af[i] = ld_afrag(&Asb[wm * 32 + i * 16 + lm][0], hf);
#pragma unroll
  for (int j = 0; j < 4; ++j) bf[j] = ld_bfrag(&Bsb[wn * 64 + j * 16 + lm][0], hf);
#pragma unroll
  for (int i = 0; i < 2; ++i)
#pragma unroll
    for (int j = 0; j < 4; ++j) acc[i][j] = wmma_bf16(af[i], bf[j], acc[i][j]);
}

__global__ __launch_bounds__(256) void gemm_bf16_nt(const __bf16* __restrict__ A,
                                                    const __bf16* __restrict__ Bt,
                                                    const float* __restrict__ bias,
                                                    float* __restrict__ Cf,
                                                    __bf16* __restrict__ Cb,
                                                    int M, int N, int K,
                                                    int mode, float scale) {
  __shared__ __bf16 As[2][64][LDSTR];
  __shared__ __bf16 Bs[2][256][LDSTR];
  const int t = threadIdx.x;
  const int wid = t >> 5, ln = t & 31, hf = ln >> 4, lm = ln & 15;
  const int wn = wid & 3, wm = wid >> 2;   // waves: 2(M) x 4(N)
  const int mbase = blockIdx.y * 64, nbase = blockIdx.x * 256;

  // Loader mapping: per async b128 instruction, 256 lanes x 16B = one 64x32 tile.
  const int lr = t >> 2, ls = (t & 3) * 8;
  const uint32_t voffA = (uint32_t)(((size_t)(mbase + lr) * K + ls) * 2);
  uint32_t voffB[4];
  uint32_t aLds[2], bLds[2][4];
#pragma unroll
  for (int j = 0; j < 4; ++j)
    voffB[j] = (uint32_t)(((size_t)(nbase + j * 64 + lr) * K + ls) * 2);
#pragma unroll
  for (int p = 0; p < 2; ++p) {
    aLds[p] = lds_addr_of(&As[p][lr][ls]);
#pragma unroll
    for (int j = 0; j < 4; ++j) bLds[p][j] = lds_addr_of(&Bs[p][j * 64 + lr][ls]);
  }

  v8f acc[2][4];
#pragma unroll
  for (int i = 0; i < 2; ++i)
#pragma unroll
    for (int j = 0; j < 4; ++j) acc[i][j] = (v8f){};

  // prologue: stage kb=0 into buffer 0
  gemm_issue(aLds[0], bLds[0], A, Bt, voffA, voffB);
  wait_async0();
  __syncthreads();

  // K is a multiple of 64 (1024 or 4096): process stages in parity pairs so
  // the double-buffer index is compile-time constant.
  for (int kb = 0; kb < K; kb += 64) {
    // stage kb+32 into buffer 1 (always in range inside the loop)
    gemm_issue(aLds[1], bLds[1], A + kb + 32, Bt + kb + 32, voffA, voffB);
    gemm_stage(As[0], Bs[0], wm, wn, lm, hf, acc);
    wait_async0();
    __syncthreads();
    if (kb + 64 < K)   // stage kb+64 into buffer 0
      gemm_issue(aLds[0], bLds[0], A + kb + 64, Bt + kb + 64, voffA, voffB);
    gemm_stage(As[1], Bs[1], wm, wn, lm, hf, acc);
    wait_async0();
    __syncthreads();
  }

#pragma unroll
  for (int i = 0; i < 2; ++i)
#pragma unroll
    for (int j = 0; j < 4; ++j) {
      const int col = nbase + wn * 64 + j * 16 + lm;
      const float bj = (mode != 0) ? bias[col] : 0.0f;
#pragma unroll
      for (int r = 0; r < 8; ++r) {
        const int row = mbase + wm * 32 + i * 16 + r + hf * 8;
        const float vv = acc[i][j][r];
        if (mode == 0) {
          Cf[(size_t)row * N + col] = vv;
        } else if (mode == 1) {
          Cb[(size_t)row * N + col] = (__bf16)((vv + bj) * scale);
        } else {
          const float xg = vv + bj;                       // fast_gelu = x*sigmoid(1.702x)
          Cb[(size_t)row * N + col] = (__bf16)(xg / (1.0f + __expf(-1.702f * xg)));
        }
      }
    }
}

// ---------------------------------------------------------------------------
// Flash attention.  Grid = B*H*4; block = 256 (8 waves); each wave owns one
// 16-row q tile of a 128-row q block.  K [512x64] and V^T [64x512] staged in
// dynamic LDS (K via async-to-LDS); online softmax over key blocks of 32.
// q is pre-scaled by 1/sqrt(dh) in its projection epilogue.
// ---------------------------------------------------------------------------
#define KS_STRIDE 72
#define VT_STRIDE 520
#define ATTN_LDS_ELEMS (512 * KS_STRIDE + 64 * VT_STRIDE + 8 * 16 * 32)

__global__ __launch_bounds__(256) void attn_flash(const __bf16* __restrict__ q,
                                                  const __bf16* __restrict__ k,
                                                  const __bf16* __restrict__ v,
                                                  const int* __restrict__ lengths,
                                                  __bf16* __restrict__ ctx) {
  extern __shared__ char smem[];
  __bf16* Ks = (__bf16*)smem;                 // [512][72]
  __bf16* Vt = Ks + 512 * KS_STRIDE;          // [64][520]
  __bf16* Pb = Vt + 64 * VT_STRIDE;           // [8 waves][16][32]

  const int t    = threadIdx.x;
  const int qblk = blockIdx.x & 3;
  const int hh   = (blockIdx.x >> 2) & 15;
  const int b    = blockIdx.x >> 6;
  const int len  = lengths[b];

  // --- stage K rows (row-major) via async copy to LDS ---
  {
    const void* kbase = (const void*)(k + ((size_t)(b * SS) * DD + hh * DH));
    for (int r = t; r < 512; r += 256) {
      const uint32_t voff = (uint32_t)((size_t)r * DD * 2);
      const uint32_t la   = lds_addr_of(&Ks[r * KS_STRIDE]);
#pragma unroll
      for (int j = 0; j < 8; ++j) async_ld_b128(la + j * 16, kbase, voff + j * 16);
    }
  }
  // --- stage V transposed (Vt[d][s]); needs a VGPR round trip for the scatter ---
  for (int r = t; r < 512; r += 256) {
    union { uint4 u[8]; __bf16 e[64]; } rowv;
    const uint4* src = (const uint4*)(v + ((size_t)(b * SS + r) * DD + hh * DH));
#pragma unroll
    for (int j = 0; j < 8; ++j) rowv.u[j] = src[j];
#pragma unroll
    for (int d = 0; d < 64; ++d) Vt[d * VT_STRIDE + r] = rowv.e[d];
  }
  wait_async0();
  __syncthreads();

  const int wid = t >> 5, ln = t & 31, hf = ln >> 4, lm = ln & 15;
  const int q0 = qblk * 128 + wid * 16;
  __bf16* Pw = Pb + wid * 512;

  // Q A-fragments for the two dh K-steps
  const __bf16* qrow = q + ((size_t)(b * SS + q0 + lm) * DD + hh * DH);
  FragU aq[2];
#pragma unroll
  for (int s = 0; s < 2; ++s) {
    aq[s].u[0] = *(const uint4*)(qrow + s * 32 + hf * 16);
    aq[s].u[1] = *(const uint4*)(qrow + s * 32 + hf * 16 + 8);
  }

  v8f acc[4] = {{}, {}, {}, {}};
  float rowM[8], rowS[8];
#pragma unroll
  for (int r = 0; r < 8; ++r) { rowM[r] = -3.0e38f; rowS[r] = 0.0f; }

  for (int kb = 0; kb < 512; kb += 32) {
    // --- score tiles for key columns [kb, kb+16) and [kb+16, kb+32) ---
    v8f s0 = {}, s1 = {};
#pragma unroll
    for (int s = 0; s < 2; ++s) {
      FragU bk0, bk1;
      const __bf16* kr0 = &Ks[(kb + lm) * KS_STRIDE];
      const __bf16* kr1 = &Ks[(kb + 16 + lm) * KS_STRIDE];
      bk0.u[0] = *(const uint4*)(kr0 + s * 32 + hf * 16);
      bk0.u[1] = *(const uint4*)(kr0 + s * 32 + hf * 16 + 8);
      bk1.u[0] = *(const uint4*)(kr1 + s * 32 + hf * 16);
      bk1.u[1] = *(const uint4*)(kr1 + s * 32 + hf * 16 + 8);
      s0 = wmma_bf16(aq[s].v, bk0.v, s0);
      s1 = wmma_bf16(aq[s].v, bk1.v, s1);
    }
    const float mk0 = (kb + lm)      < len ? 0.0f : -1e9f;
    const float mk1 = (kb + 16 + lm) < len ? 0.0f : -1e9f;

    // --- online softmax update (row m = r + 8*hf lives on one 16-lane half) ---
#pragma unroll
    for (int r = 0; r < 8; ++r) {
      const float sa = s0[r] + mk0;
      const float sb = s1[r] + mk1;
      float m = fmaxf(sa, sb);
#pragma unroll
      for (int msk = 8; msk; msk >>= 1) m = fmaxf(m, __shfl_xor(m, msk, 32));
      const float newM = fmaxf(rowM[r], m);
      const float cs   = __expf(rowM[r] - newM);
      rowM[r] = newM;
      const float p0 = __expf(sa - newM);
      const float p1 = __expf(sb - newM);
      float ps = p0 + p1;
#pragma unroll
      for (int msk = 8; msk; msk >>= 1) ps += __shfl_xor(ps, msk, 32);
      rowS[r] = rowS[r] * cs + ps;
#pragma unroll
      for (int n = 0; n < 4; ++n) acc[n][r] *= cs;
      // bounce P through LDS to re-layout C-fragment -> A-fragment
      Pw[(r + hf * 8) * 32 + lm]      = (__bf16)p0;
      Pw[(r + hf * 8) * 32 + 16 + lm] = (__bf16)p1;
    }

    // --- ctx += P(16x32) x V(32x64) ---  (same-wave LDS ops stay in order)
    FragU ap;
    ap.u[0] = *(const uint4*)&Pw[lm * 32 + hf * 16];
    ap.u[1] = *(const uint4*)&Pw[lm * 32 + hf * 16 + 8];
#pragma unroll
    for (int n = 0; n < 4; ++n) {
      FragU bv;
      const __bf16* vr = &Vt[(n * 16 + lm) * VT_STRIDE + kb];
      bv.u[0] = *(const uint4*)(vr + hf * 16);
      bv.u[1] = *(const uint4*)(vr + hf * 16 + 8);
      acc[n] = wmma_bf16(ap.v, bv.v, acc[n]);
    }
  }

  // --- normalize and emit ctx (bf16, feeds the Wo GEMM) ---
#pragma unroll
  for (int n = 0; n < 4; ++n)
#pragma unroll
    for (int r = 0; r < 8; ++r) {
      const int row = q0 + r + hf * 8;
      const int col = hh * DH + n * 16 + lm;
      ctx[(size_t)(b * SS + row) * DD + col] = (__bf16)(acc[n][r] / rowS[r]);
    }
}

// ---------------------------------------------------------------------------
// Host orchestration
// ---------------------------------------------------------------------------
extern "C" void kernel_launch(void* const* d_in, const int* in_sizes, int n_in,
                              void* d_out, int out_size, void* d_ws, size_t ws_size,
                              hipStream_t stream) {
  (void)in_sizes; (void)n_in; (void)out_size; (void)ws_size;
  const int*   x        = (const int*)d_in[0];
  const int*   lengths  = (const int*)d_in[1];
  const float* word_emb = (const float*)d_in[2];
  const float* pos_emb  = (const float*)d_in[3];
  const float* ln_emb_g = (const float*)d_in[4];
  const float* ln_emb_b = (const float*)d_in[5];
  const float* Wq = (const float*)d_in[6];   const float* bq = (const float*)d_in[7];
  const float* Wk = (const float*)d_in[8];   const float* bk = (const float*)d_in[9];
  const float* Wv = (const float*)d_in[10];  const float* bv = (const float*)d_in[11];
  const float* Wo = (const float*)d_in[12];  const float* bo = (const float*)d_in[13];
  const float* ln1_g = (const float*)d_in[14]; const float* ln1_b = (const float*)d_in[15];
  const float* W1 = (const float*)d_in[16];  const float* b1 = (const float*)d_in[17];
  const float* W2 = (const float*)d_in[18];  const float* b2 = (const float*)d_in[19];
  const float* ln2_g = (const float*)d_in[20]; const float* ln2_b = (const float*)d_in[21];

  char* ws = (char*)d_ws;
  size_t off = 0;
  auto alloc = [&](size_t bytes) -> void* {
    void* p = ws + off;
    off += (bytes + 255) & ~(size_t)255;
    return p;
  };

  __bf16 *wtq[LL], *wtk[LL], *wtv[LL], *wto[LL], *wt1[LL], *wt2[LL];
  for (int i = 0; i < LL; ++i) {
    wtq[i] = (__bf16*)alloc((size_t)DD * DD * 2);
    wtk[i] = (__bf16*)alloc((size_t)DD * DD * 2);
    wtv[i] = (__bf16*)alloc((size_t)DD * DD * 2);
    wto[i] = (__bf16*)alloc((size_t)DD * DD * 2);
    wt1[i] = (__bf16*)alloc((size_t)FF * DD * 2);   // W1^T [F,D]
    wt2[i] = (__bf16*)alloc((size_t)DD * FF * 2);   // W2^T [D,F]
  }
  float*  h    = (float*)alloc((size_t)MM * DD * 4);
  __bf16* hb   = (__bf16*)alloc((size_t)MM * DD * 2);
  __bf16* qb   = (__bf16*)alloc((size_t)MM * DD * 2);
  __bf16* kb2  = (__bf16*)alloc((size_t)MM * DD * 2);
  __bf16* vb2  = (__bf16*)alloc((size_t)MM * DD * 2);
  __bf16* ctxb = (__bf16*)alloc((size_t)MM * DD * 2);
  float*  raw  = (float*)alloc((size_t)MM * DD * 4);
  __bf16* ff1  = (__bf16*)alloc((size_t)MM * FF * 2);

  // --- weight transpose/convert (runs every call; deterministic) ---
  const dim3 tb(32, 8);
  for (int i = 0; i < LL; ++i) {
    transpose_cvt<<<dim3(DD / 32, DD / 32), tb, 0, stream>>>(Wq + (size_t)i * DD * DD, wtq[i], DD, DD);
    transpose_cvt<<<dim3(DD / 32, DD / 32), tb, 0, stream>>>(Wk + (size_t)i * DD * DD, wtk[i], DD, DD);
    transpose_cvt<<<dim3(DD / 32, DD / 32), tb, 0, stream>>>(Wv + (size_t)i * DD * DD, wtv[i], DD, DD);
    transpose_cvt<<<dim3(DD / 32, DD / 32), tb, 0, stream>>>(Wo + (size_t)i * DD * DD, wto[i], DD, DD);
    transpose_cvt<<<dim3(FF / 32, DD / 32), tb, 0, stream>>>(W1 + (size_t)i * DD * FF, wt1[i], DD, FF);
    transpose_cvt<<<dim3(DD / 32, FF / 32), tb, 0, stream>>>(W2 + (size_t)i * FF * DD, wt2[i], FF, DD);
  }

  embed_ln<<<MM, 256, 0, stream>>>(x, lengths, word_emb, pos_emb, ln_emb_g, ln_emb_b, h, hb);

  const size_t attn_smem = (size_t)ATTN_LDS_ELEMS * sizeof(__bf16);
  (void)hipFuncSetAttribute(reinterpret_cast<const void*>(attn_flash),
                            hipFuncAttributeMaxDynamicSharedMemorySize, (int)attn_smem);

  const float qscale = 0.125f;   // 1/sqrt(dh)
  for (int i = 0; i < LL; ++i) {
    // Q/K/V projections (bias fused; Q pre-scaled)
    gemm_bf16_nt<<<dim3(DD / 256, MM / 64), 256, 0, stream>>>(
        hb, wtq[i], bq + (size_t)i * DD, nullptr, qb, MM, DD, DD, 1, qscale);
    gemm_bf16_nt<<<dim3(DD / 256, MM / 64), 256, 0, stream>>>(
        hb, wtk[i], bk + (size_t)i * DD, nullptr, kb2, MM, DD, DD, 1, 1.0f);
    gemm_bf16_nt<<<dim3(DD / 256, MM / 64), 256, 0, stream>>>(
        hb, wtv[i], bv + (size_t)i * DD, nullptr, vb2, MM, DD, DD, 1, 1.0f);
    // attention
    attn_flash<<<BB * HH * 4, 256, attn_smem, stream>>>(qb, kb2, vb2, lengths, ctxb);
    // output projection (raw f32; bias folded into LN kernel)
    gemm_bf16_nt<<<dim3(DD / 256, MM / 64), 256, 0, stream>>>(
        ctxb, wto[i], nullptr, raw, nullptr, MM, DD, DD, 0, 1.0f);
    ln_residual<<<MM, 256, 0, stream>>>(h, raw, bo + (size_t)i * DD,
                                        ln1_g + (size_t)i * DD, ln1_b + (size_t)i * DD,
                                        lengths, 0, h, hb);
    // FFN
    gemm_bf16_nt<<<dim3(FF / 256, MM / 64), 256, 0, stream>>>(
        hb, wt1[i], b1 + (size_t)i * FF, nullptr, ff1, MM, FF, DD, 2, 1.0f);
    gemm_bf16_nt<<<dim3(DD / 256, MM / 64), 256, 0, stream>>>(
        ff1, wt2[i], nullptr, raw, nullptr, MM, DD, FF, 0, 1.0f);
    float* outf = (i == LL - 1) ? (float*)d_out : h;
    ln_residual<<<MM, 256, 0, stream>>>(h, raw, b2 + (size_t)i * DD,
                                        ln2_g + (size_t)i * DD, ln2_b + (size_t)i * DD,
                                        lengths, 1, outf, hb);
  }
}
